// TransportKernel_37203006718368
// MI455X (gfx1250) — compile-verified, matrix operand order checked by reading, not run on hardware
//
#include <hip/hip_runtime.h>

typedef __attribute__((ext_vector_type(16))) _Float16 v16h;
typedef __attribute__((ext_vector_type(2)))  _Float16 h2;
typedef __attribute__((ext_vector_type(8)))  float    v8f;
typedef __attribute__((ext_vector_type(8)))  unsigned v8u;

#if __has_builtin(__builtin_amdgcn_exp2f)
#define EXP2F(x) __builtin_amdgcn_exp2f(x)
#else
#define EXP2F(x) exp2f(x)
#endif

// exp(-d2/2)  = exp2(d2 * -0.5*log2(e))
// exp(-d2/8)  = exp2(d2 * -0.125*log2(e))
#define C_FIT (-0.72134752044448170368f)
#define C_MMD (-0.18033688011112042592f)
#define DD 16

__device__ __forceinline__ unsigned pkh(float a, float b) {
    h2 h;
    h[0] = (_Float16)a;
    h[1] = (_Float16)b;
    return __builtin_bit_cast(unsigned, h);
}

__global__ void tk_zero(float* out, int n) {
    int i = blockIdx.x * blockDim.x + threadIdx.x;
    if (i < n) out[i] = 0.0f;
}

// Per row i: f16 copy of X, transposed f16 Lam, row squared-norms of X.
__global__ void tk_prep(const float* __restrict__ X, const float* __restrict__ Lam,
                        _Float16* __restrict__ Xh, _Float16* __restrict__ LamT,
                        float* __restrict__ nrm, int n) {
    int i = blockIdx.x * blockDim.x + threadIdx.x;
    if (i >= n) return;
    float s = 0.0f;
#pragma unroll
    for (int d = 0; d < DD; ++d) {
        float x = X[i * DD + d];
        s += x * x;
        Xh[i * DD + d] = (_Float16)x;
        LamT[d * n + i] = (_Float16)Lam[i * DD + d];
    }
    nrm[i] = s;
}

// One wave owns one 16-row i-tile; loops over j-tiles two at a time.
// Pass A (MODE=0): G = K1 @ Lam; writes diff = G - Y (f32 + f16 transposed),
//                  accumulates 0.01 * sum(Lam * G) into out.
// Pass B (MODE=1): H = K2 @ diff; accumulates sum(diff * H) into out.
template <int MODE>
__global__ void tk_gram_pass(const _Float16* __restrict__ Xh,
                             const _Float16* __restrict__ Bt,   // LamT (A) or diffT (B)
                             const float* __restrict__ nrm,
                             const float* __restrict__ Lam32,   // pass A only
                             const float* __restrict__ Y,       // pass A only
                             float* __restrict__ diffW,         // f32 diff (A: write, B: read)
                             _Float16* __restrict__ diffT,      // pass A only
                             float* __restrict__ out, int n) {
    const int lane     = threadIdx.x & 31;
    const int wid      = blockIdx.x * (blockDim.x >> 5) + (threadIdx.x >> 5);
    const int iBase    = wid * 16;
    const int hi       = lane >> 4;      // 0 = lanes 0-15, 1 = lanes 16-31
    const int lo       = lane & 15;
    const bool lowHalf = (lane < 16);
    const float coef   = (MODE == 0) ? C_FIT : C_MMD;

    // Loop-invariant B operand of WMMA #1: column n=lo of X_i^T == row (iBase+lo)
    // of Xh in lanes 0-15. Upper lanes (K=16..31) are ZERO here, which lets the
    // per-iteration A operand be loaded by ALL lanes with no predication.
    v16h bI;
    if (lowHalf) {
        bI = *(const v16h*)(Xh + (iBase + lo) * DD);
    } else {
#pragma unroll
        for (int k = 0; k < 16; ++k) bI[k] = (_Float16)0.0f;
    }
    const float ni = nrm[iBase + lo];

    v8f acc = {};                        // C-layout: acc[r]@lane = G[iBase+r+8*hi][d=lo]
    const int nTiles = n >> 4;
    for (int jt = 0; jt < nTiles; jt += 2) {
        const int jBase = jt << 4;       // tile0 = jBase, tile1 = jBase+16

        // WMMA #1 (x2): S = X_j . X_i^T.  All lanes load (upper-lane garbage is
        // annihilated by bI's zeroed K=16..31 rows).
        v16h aJ0 = *(const v16h*)(Xh + (jBase + lo) * DD);
        v16h aJ1 = *(const v16h*)(Xh + (jBase + 16 + lo) * DD);
        v8f zc = {};
        v8f s0 = __builtin_amdgcn_wmma_f32_16x16x32_f16(false, aJ0, false, bI,
                                                        (short)0, zc, false, false);
        v8f s1 = __builtin_amdgcn_wmma_f32_16x16x32_f16(false, aJ1, false, bI,
                                                        (short)0, zc, false, false);

        // d2 -> k = exp2(coef*d2).  t*[r]@lane = K[j = base + r + 8*hi][i = iBase+lo]
        float t0[8], t1[8];
#pragma unroll
        for (int r = 0; r < 8; ++r) {
            float nj0 = nrm[jBase + 8 * hi + r];
            float nj1 = nrm[jBase + 16 + 8 * hi + r];
            t0[r] = EXP2F(coef * fmaxf(ni + nj0 - 2.0f * s0[r], 0.0f));
            t1[r] = EXP2F(coef * fmaxf(ni + nj1 - 2.0f * s1[r], 0.0f));
        }

        // Build A operand of WMMA #2 with BOTH tiles: K=0..15 <- tile0, K=16..31 <- tile1.
        // Pack to f16 pairs first so the half-swap shuffle moves 32-bit packs.
        unsigned p0[4], p1[4], q0[4], q1[4];
#pragma unroll
        for (int p = 0; p < 4; ++p) {
            p0[p] = pkh(t0[2 * p], t0[2 * p + 1]);
            p1[p] = pkh(t1[2 * p], t1[2 * p + 1]);
            q0[p] = __shfl_xor(p0[p], 16, 32);
            q1[p] = __shfl_xor(p1[p], 16, 32);
        }
        v8u au;
#pragma unroll
        for (int p = 0; p < 4; ++p) {
            au[p]     = hi ? q1[p] : p0[p];  // K(local) 0..7
            au[p + 4] = hi ? p1[p] : q0[p];  // K(local) 8..15
        }
        v16h a2 = __builtin_bit_cast(v16h, au);

        // B operand of WMMA #2: lanes 0-15 take tile0 rows, lanes 16-31 tile1 rows.
        v16h b2 = *(const v16h*)(Bt + lo * n + jBase + hi * 16);

        acc = __builtin_amdgcn_wmma_f32_16x16x32_f16(false, a2, false, b2,
                                                     (short)0, acc, false, false);
    }

    // Epilogue
    float part = 0.0f;
#pragma unroll
    for (int r = 0; r < 8; ++r) {
        const int gi  = iBase + r + 8 * hi;
        const int idx = gi * DD + lo;
        const float v = acc[r];
        if (MODE == 0) {
            part += Lam32[idx] * v;           // reg term uses G = K1@Lam
            const float dv = v - Y[idx];
            diffW[idx] = dv;
            diffT[lo * n + gi] = (_Float16)dv;
        } else {
            part += diffW[idx] * v;           // fit term: diff . (K2@diff)
        }
    }
#pragma unroll
    for (int off = 16; off > 0; off >>= 1) part += __shfl_xor(part, off, 32);
    if (lane == 0) atomicAdd(out, (MODE == 0) ? 0.01f * part : part);
}

extern "C" void kernel_launch(void* const* d_in, const int* in_sizes, int n_in,
                              void* d_out, int out_size, void* d_ws, size_t ws_size,
                              hipStream_t stream) {
    const float* X   = (const float*)d_in[0];
    const float* Y   = (const float*)d_in[1];
    const float* Lam = (const float*)d_in[2];
    float* out = (float*)d_out;
    const int n = in_sizes[0] / DD;   // 8192

    // Workspace layout (all offsets 256B-aligned); total ~1.32 MB.
    char* w = (char*)d_ws;
    _Float16* Xh    = (_Float16*)(w);                         // n*16*2 = 256 KB
    _Float16* LamT  = (_Float16*)(w + 0x40000);               // 256 KB
    _Float16* diffT = (_Float16*)(w + 0x80000);               // 256 KB
    float*    nrm   = (float*)   (w + 0xC0000);               // n*4   =  32 KB
    float*    diffW = (float*)   (w + 0xD0000);               // n*16*4= 512 KB

    tk_zero<<<1, 64, 0, stream>>>(out, out_size);
    tk_prep<<<(n + 255) / 256, 256, 0, stream>>>(X, Lam, Xh, LamT, nrm, n);

    const int nTiles = n / 16;                 // 512 i-tiles, one wave each
    const int wavesPerBlock = 4;               // 128 threads/block
    dim3 blk(32 * wavesPerBlock);
    dim3 grd(nTiles / wavesPerBlock);
    tk_gram_pass<0><<<grd, blk, 0, stream>>>(Xh, LamT, nrm, Lam, Y, diffW, diffT, out, n);
    tk_gram_pass<1><<<grd, blk, 0, stream>>>(Xh, diffT, nrm, nullptr, nullptr, diffW, nullptr, out, n);
}
